// TransformerLayer_9028021256840
// MI455X (gfx1250) — matrix operand order repaired and checked by reading, not compiled, not run on hardware
//
#include <hip/hip_runtime.h>
#include <hip/hip_bf16.h>

typedef _Float16 half8  __attribute__((ext_vector_type(8)));
typedef _Float16 half16 __attribute__((ext_vector_type(16)));
typedef float    float8 __attribute__((ext_vector_type(8)));

#define LN2F       0.6931471805599453f
#define INV_SQRT_D 0.17677669529663687f   /* 1/sqrt(32) */

__device__ __forceinline__ half16 cat8(half8 lo, half8 hi) {
  return __builtin_shufflevector(lo, hi, 0,1,2,3,4,5,6,7,8,9,10,11,12,13,14,15);
}

// ---------------------------------------------------------------- utilities
__global__ void fill_f32_kernel(float* __restrict__ p, float v, int n) {
  int i = blockIdx.x * blockDim.x + threadIdx.x;
  if (i < n) p[i] = v;
}

// Pack f32 weight [K,128] (row-major, [in,out]) into f16 WMMA-B fragment order:
//   dst[((s*8 + ct)*32 + lane)*16 + h] = W[(s*32 + g*16 + h)*128 + ct*16 + n]
// with lane = g*16 + n.  Each lane's 16 fragment halves become contiguous (32B).
__global__ void pack_b_kernel(const float* __restrict__ src,
                              _Float16* __restrict__ dst, int K)
{
  int i = blockIdx.x * blockDim.x + threadIdx.x;
  if (i >= K * 128) return;
  int h  = i & 15;
  int l  = (i >> 4) & 31;
  int ct = (i >> 9) & 7;
  int s  = i >> 12;
  int n = l & 15, g = l >> 4;
  dst[i] = (_Float16)src[(size_t)(s * 32 + g * 16 + h) * 128 + ct * 16 + n];
}

// --------------------------------------------------------------- layer norm
struct H4 { _Float16 v[4]; };

__global__ __launch_bounds__(256) void ln_kernel(
    const float* __restrict__ x, const float* __restrict__ sc,
    const float* __restrict__ bi, _Float16* __restrict__ out, int N)
{
  int row  = blockIdx.x * 8 + (threadIdx.x >> 5);
  int lane = threadIdx.x & 31;
  if (row >= N) return;                         // wave-uniform exit
  const float* xr = x + (size_t)row * 128 + lane * 4;
  float4 v = *(const float4*)xr;
  float s  = v.x + v.y + v.z + v.w;
  float s2 = v.x*v.x + v.y*v.y + v.z*v.z + v.w*v.w;
  #pragma unroll
  for (int o = 1; o < 32; o <<= 1) {
    s  += __shfl_xor(s,  o, 32);
    s2 += __shfl_xor(s2, o, 32);
  }
  float mean = s * (1.f / 128.f);
  float var  = s2 * (1.f / 128.f) - mean * mean;
  float rs   = rsqrtf(var + 1e-5f);
  int c = lane * 4;
  H4 o4;
  o4.v[0] = (_Float16)((v.x - mean) * rs * sc[c+0] + bi[c+0]);
  o4.v[1] = (_Float16)((v.y - mean) * rs * sc[c+1] + bi[c+1]);
  o4.v[2] = (_Float16)((v.z - mean) * rs * sc[c+2] + bi[c+2]);
  o4.v[3] = (_Float16)((v.w - mean) * rs * sc[c+3] + bi[c+3]);
  *(H4*)(out + (size_t)row * 128 + c) = o4;
}

// -------------------------------------------------- generic 128-K WMMA GEMM
#define RT_PER 8
template <bool HAS_BIAS, int ACT, bool HAS_RES, bool STORE_HALF>
__global__ __launch_bounds__(32) void gemm16_kernel(
    const _Float16* __restrict__ A, const _Float16* __restrict__ Bp,
    const float* __restrict__ bias, const float* __restrict__ res,
    float* __restrict__ outF, _Float16* __restrict__ outH, int M)
{
  const int lane = threadIdx.x;
  const int q = lane & 15, g = lane >> 4;
  const int ct = blockIdx.x;                 // column tile 0..7

  half16 bf[4];
  #pragma unroll
  for (int s = 0; s < 4; ++s)
    bf[s] = *(const half16*)(Bp + ((size_t)(s * 8 + ct) * 32 + lane) * 16);

  const int cc = ct * 16 + q;
  const float bv = HAS_BIAS ? bias[cc] : 0.f;

  for (int t = 0; t < RT_PER; ++t) {
    int rt = blockIdx.y * RT_PER + t;
    if (rt * 16 >= M) break;                 // block-uniform
    const bool full = (rt * 16 + 16) <= M;   // block-uniform fast path

    int arow = rt * 16 + q;
    if (!full && arow >= M) arow = M - 1;
    const _Float16* Ar = A + (size_t)arow * 128;

    half16 a[4];
    #pragma unroll
    for (int s = 0; s < 4; ++s) {
      half8 lo = *(const half8*)(Ar + s * 32 + g * 16);
      half8 hi = *(const half8*)(Ar + s * 32 + g * 16 + 8);
      a[s] = cat8(lo, hi);
    }

    float8 c = {};
    #pragma unroll
    for (int s = 0; s < 4; ++s)
      c = __builtin_amdgcn_wmma_f32_16x16x32_f16(false, a[s], false, bf[s],
                                                 (short)0, c, false, false);

    if (full) {
      #pragma unroll
      for (int v = 0; v < 8; ++v) {
        size_t r = (size_t)(rt * 16 + v + 8 * g);
        float val = c[v] + bv;
        if (ACT == 1) val = fmaxf(val, 0.f);
        if (HAS_RES) val += res[r * 128 + cc];
        if (STORE_HALF) outH[r * 128 + cc] = (_Float16)val;
        else            outF[r * 128 + cc] = val;
      }
    } else {
      #pragma unroll
      for (int v = 0; v < 8; ++v) {
        int r = rt * 16 + v + 8 * g;
        if (r < M) {
          float val = c[v] + bv;
          if (ACT == 1) val = fmaxf(val, 0.f);
          if (HAS_RES) val += res[(size_t)r * 128 + cc];
          if (STORE_HALF) outH[(size_t)r * 128 + cc] = (_Float16)val;
          else            outF[(size_t)r * 128 + cc] = val;
        }
      }
    }
  }
}

// ------------------------------------------- edge K/V GEMM + logits kernel
__global__ __launch_bounds__(256) void edge_kv_kernel(
    const float* __restrict__ hE, const _Float16* __restrict__ y16,
    const int* __restrict__ eidx, const float* __restrict__ Q,
    const _Float16* __restrict__ WKp, const _Float16* __restrict__ WVp,
    float* __restrict__ logits, _Float16* __restrict__ V16,
    int E, int numTiles)
{
  __shared__ _Float16 lds[16][264];           // 16 x 256 he tile, padded rows
  const int tid  = threadIdx.x;
  const int lane = tid & 31;
  const int ct   = tid >> 5;                  // wave index == column tile
  const int q = lane & 15, g = lane >> 4;

  half16 bk[8], bv[8];
  #pragma unroll
  for (int s = 0; s < 8; ++s) {
    size_t off = ((size_t)(s * 8 + ct) * 32 + lane) * 16;
    bk[s] = *(const half16*)(WKp + off);
    bv[s] = *(const half16*)(WVp + off);
  }

  for (int tile = blockIdx.x; tile < numTiles; tile += gridDim.x) {
    const bool full = (tile * 16 + 16) <= E;  // block-uniform
    { // cooperative staging of he = [f16(h_E[e]) | y16[col[e]]]
      int r = tid >> 4, seg = tid & 15;
      int e = tile * 16 + r; if (e >= E) e = E - 1;
      const float* src = hE + (size_t)e * 128 + seg * 8;
      float4 f0 = *(const float4*)src;
      float4 f1 = *(const float4*)(src + 4);
      half8 hv;
      hv[0]=(_Float16)f0.x; hv[1]=(_Float16)f0.y; hv[2]=(_Float16)f0.z; hv[3]=(_Float16)f0.w;
      hv[4]=(_Float16)f1.x; hv[5]=(_Float16)f1.y; hv[6]=(_Float16)f1.z; hv[7]=(_Float16)f1.w;
      *(half8*)&lds[r][seg * 8] = hv;
      int col = eidx[E + e];
      *(half8*)&lds[r][128 + seg * 8] =
          *(const half8*)(y16 + (size_t)col * 128 + seg * 8);
      // prefetch next grid-stride tile of the HBM-streaming operand
      int tile2 = tile + gridDim.x;
      if (tile2 < numTiles) {
        int e2 = tile2 * 16 + r;
        if (e2 < E) __builtin_prefetch(hE + (size_t)e2 * 128 + seg * 8, 0, 1);
      }
    }
    __syncthreads();

    float8 ck = {}, cv = {};
    const _Float16* Ar = &lds[q][0];
    #pragma unroll
    for (int sg = 0; sg < 2; ++sg) {
      half16 a[4];
      #pragma unroll
      for (int si = 0; si < 4; ++si) {
        int s = sg * 4 + si;
        half8 lo = *(const half8*)(Ar + s * 32 + g * 16);
        half8 hi = *(const half8*)(Ar + s * 32 + g * 16 + 8);
        a[si] = cat8(lo, hi);
      }
      #pragma unroll
      for (int si = 0; si < 4; ++si) {
        int s = sg * 4 + si;
        ck = __builtin_amdgcn_wmma_f32_16x16x32_f16(false, a[si], false, bk[s],
                                                    (short)0, ck, false, false);
        cv = __builtin_amdgcn_wmma_f32_16x16x32_f16(false, a[si], false, bv[s],
                                                    (short)0, cv, false, false);
      }
    }

    const int head = ct >> 1;
    if (full) {
      // V stores, unguarded
      #pragma unroll
      for (int v = 0; v < 8; ++v) {
        size_t e = (size_t)(tile * 16 + v + 8 * g);
        V16[e * 128 + ct * 16 + q] = (_Float16)cv[v];
      }
      // hoisted gathers (clause-friendly)
      int rows[8];
      #pragma unroll
      for (int v = 0; v < 8; ++v) rows[v] = eidx[tile * 16 + v + 8 * g];
      float qv[8];
      #pragma unroll
      for (int v = 0; v < 8; ++v)
        qv[v] = Q[(size_t)rows[v] * 128 + ct * 16 + q];
      // per-v reduce; lane q==v keeps the sum -> one atomic for all 8 values
      float myp = 0.f;
      #pragma unroll
      for (int v = 0; v < 8; ++v) {
        float p = ck[v] * qv[v];
        p += __shfl_xor(p, 1, 32);
        p += __shfl_xor(p, 2, 32);
        p += __shfl_xor(p, 4, 32);
        p += __shfl_xor(p, 8, 32);
        if (q == v) myp = p;                 // v_cndmask select
      }
      if (q < 8) {
        size_t e = (size_t)(tile * 16 + q + 8 * g);
        atomicAdd(&logits[e * 4 + head], myp);
      }
    } else {
      #pragma unroll
      for (int v = 0; v < 8; ++v) {
        int et = v + 8 * g;
        int e  = tile * 16 + et;
        bool ok = (e < E); if (!ok) e = E - 1;
        if (ok) V16[(size_t)e * 128 + ct * 16 + q] = (_Float16)cv[v];
        int rowN = eidx[e];
        float p = ck[v] * Q[(size_t)rowN * 128 + ct * 16 + q];
        p += __shfl_xor(p, 1, 32);
        p += __shfl_xor(p, 2, 32);
        p += __shfl_xor(p, 4, 32);
        p += __shfl_xor(p, 8, 32);
        if (ok && q == 0) atomicAdd(&logits[(size_t)e * 4 + head], p);
      }
    }
    __syncthreads();
  }
}

// ---------------------------------------------------- scatter softmax parts
__global__ void seg_max_kernel(const float* __restrict__ logits,
                               const int* __restrict__ eidx,
                               unsigned* __restrict__ mkeys, int E)
{
  int i = blockIdx.x * blockDim.x + threadIdx.x;
  if (i >= E * 4) return;
  int e = i >> 2, h = i & 3;
  int row = eidx[e];
  unsigned b = __float_as_uint(logits[i]);
  unsigned key = (b & 0x80000000u) ? ~b : (b | 0x80000000u);  // order-preserving
  atomicMax(&mkeys[row * 4 + h], key);
}

__global__ void exp_sum_kernel(float* __restrict__ logits,
                               const int* __restrict__ eidx,
                               const unsigned* __restrict__ mkeys,
                               float* __restrict__ denom, int E)
{
  int i = blockIdx.x * blockDim.x + threadIdx.x;
  if (i >= E * 4) return;
  int e = i >> 2, h = i & 3;
  int row = eidx[e];
  unsigned k = mkeys[row * 4 + h];
  float m = (k & 0x80000000u) ? __uint_as_float(k & 0x7FFFFFFFu) : __uint_as_float(~k);
  float ex = __expf(logits[i] - m);
  logits[i] = ex;                              // in-place: logits -> ex
  atomicAdd(&denom[row * 4 + h], ex);
}

__global__ void aggr_kernel(const float* __restrict__ ex,
                            const int* __restrict__ eidx,
                            const float* __restrict__ denom,
                            const _Float16* __restrict__ V16,
                            float* __restrict__ aggr, int E)
{
  int i = blockIdx.x * blockDim.x + threadIdx.x;
  if (i >= E * 4) return;
  int e = i >> 2, h = i & 3;
  int row = eidx[e];
  float alpha = ex[i] / denom[row * 4 + h] * INV_SQRT_D;
  const _Float16* vrow = V16 + (size_t)e * 128 + h * 32;
  float* arow = aggr + (size_t)row * 128 + h * 32;
  #pragma unroll 8
  for (int j = 0; j < 32; ++j)
    atomicAdd(&arow[j], alpha * (float)vrow[j]);
}

__global__ void softplus_kernel(const float* __restrict__ x,
                                _Float16* __restrict__ out, int n)
{
  int i = blockIdx.x * blockDim.x + threadIdx.x;
  if (i >= n) return;
  float v  = x[i];
  float sp = fmaxf(v, 0.f) + log1pf(__expf(-fabsf(v))) - LN2F;
  out[i] = (_Float16)sp;
}

// ------------------------------------------------------------------- driver
extern "C" void kernel_launch(void* const* d_in, const int* in_sizes, int n_in,
                              void* d_out, int out_size, void* d_ws, size_t ws_size,
                              hipStream_t stream) {
  const int H = 128;
  const int N = in_sizes[0] / H;
  const int E = in_sizes[1] / H;

  const float* hV   = (const float*)d_in[0];
  const float* hE   = (const float*)d_in[1];
  const int*   eidx = (const int*)  d_in[2];
  const float* WQ   = (const float*)d_in[3];
  const float* WK   = (const float*)d_in[4];
  const float* WV   = (const float*)d_in[5];
  const float* WO   = (const float*)d_in[6];
  const float* ln1s = (const float*)d_in[7];
  const float* ln1b = (const float*)d_in[8];
  const float* ln2s = (const float*)d_in[9];
  const float* ln2b = (const float*)d_in[10];
  const float* Win  = (const float*)d_in[11];
  const float* bin  = (const float*)d_in[12];
  const float* Wout = (const float*)d_in[13];
  const float* bout = (const float*)d_in[14];
  float* out = (float*)d_out;

  // workspace carve
  char* w = (char*)d_ws;
  auto carve = [&](size_t bytes) -> char* {
    char* p = w; w += (bytes + 255) & ~(size_t)255; return p;
  };
  _Float16* y16    = (_Float16*)carve((size_t)N * H * 2);  // y (LN1), later z (LN2)
  _Float16* s16    = (_Float16*)carve((size_t)N * H * 2);  // softplus(aggr), later t (FFN mid)
  float*    Qb     = (float*)   carve((size_t)N * H * 4);
  float*    hv1    = (float*)   carve((size_t)N * H * 4);
  float*    aggr   = (float*)   carve((size_t)N * H * 4);
  float*    mkeys  = (float*)   carve((size_t)N * 4 * 4);
  float*    denom  = (float*)   carve((size_t)N * 4 * 4);
  float*    logits = (float*)   carve((size_t)E * 4 * 4);
  _Float16* V16    = (_Float16*)carve((size_t)E * H * 2);
  _Float16* WQ16   = (_Float16*)carve(16384 * 2);
  _Float16* WK16   = (_Float16*)carve(32768 * 2);
  _Float16* WV16   = (_Float16*)carve(32768 * 2);
  _Float16* WO16   = (_Float16*)carve(16384 * 2);
  _Float16* Win16  = (_Float16*)carve(16384 * 2);
  _Float16* Wout16 = (_Float16*)carve(16384 * 2);

  auto cdiv = [](int a, int b) { return (a + b - 1) / b; };
  const dim3 ggrid(8, cdiv(N, 16 * RT_PER));

  // 1) weights -> f16, packed in WMMA B-fragment order
  pack_b_kernel<<<cdiv(16384,256),256,0,stream>>>(WQ,  WQ16,  128);
  pack_b_kernel<<<cdiv(32768,256),256,0,stream>>>(WK,  WK16,  256);
  pack_b_kernel<<<cdiv(32768,256),256,0,stream>>>(WV,  WV16,  256);
  pack_b_kernel<<<cdiv(16384,256),256,0,stream>>>(WO,  WO16,  128);
  pack_b_kernel<<<cdiv(16384,256),256,0,stream>>>(Win, Win16, 128);
  pack_b_kernel<<<cdiv(16384,256),256,0,stream>>>(Wout,Wout16,128);

  // 2) init accumulators (mkeys=0 is below every encoded float key)
  fill_f32_kernel<<<cdiv(N*4,256),  256,0,stream>>>(mkeys, 0.f, N*4);
  fill_f32_kernel<<<cdiv(N*4,256),  256,0,stream>>>(denom, 0.f, N*4);
  fill_f32_kernel<<<cdiv(N*H,256),  256,0,stream>>>(aggr,  0.f, N*H);
  fill_f32_kernel<<<cdiv(E*4,256),  256,0,stream>>>(logits,0.f, E*4);

  // 3) LN1 -> y16 ; Q = y16 @ WQ
  ln_kernel<<<cdiv(N,8),256,0,stream>>>(hV, ln1s, ln1b, y16, N);
  gemm16_kernel<false,0,false,false><<<ggrid,32,0,stream>>>(
      y16, WQ16, nullptr, nullptr, Qb, nullptr, N);

  // 4) edge K/V WMMA + logits
  int numTiles = cdiv(E, 16);
  int nBlk = numTiles < 8192 ? numTiles : 8192;
  edge_kv_kernel<<<nBlk,256,0,stream>>>(hE, y16, eidx, Qb, WK16, WV16,
                                        logits, V16, E, numTiles);

  // 5) scatter softmax + aggregation
  seg_max_kernel<<<cdiv(E*4,256),256,0,stream>>>(logits, eidx, (unsigned*)mkeys, E);
  exp_sum_kernel<<<cdiv(E*4,256),256,0,stream>>>(logits, eidx, (const unsigned*)mkeys, denom, E);
  aggr_kernel  <<<cdiv(E*4,256),256,0,stream>>>(logits, eidx, denom, V16, aggr, E);

  // 6) attn_out = softplus(aggr) @ WO ; hv1 = hV + attn_out
  softplus_kernel<<<cdiv(N*H,256),256,0,stream>>>(aggr, s16, N*H);
  gemm16_kernel<false,0,true,false><<<ggrid,32,0,stream>>>(
      s16, WO16, nullptr, hV, hv1, nullptr, N);

  // 7) FFN: z = LN2(hv1); t = relu(z@Win + bin); out = hv1 + t@Wout + bout
  ln_kernel<<<cdiv(N,8),256,0,stream>>>(hv1, ln2s, ln2b, y16, N);
  gemm16_kernel<true,1,false,true><<<ggrid,32,0,stream>>>(
      y16, Win16, bin, nullptr, nullptr, s16, N);
  gemm16_kernel<true,0,true,false><<<ggrid,32,0,stream>>>(
      s16, Wout16, bout, hv1, out, nullptr, N);
}
